// NGCFMModel_28037546508681
// MI455X (gfx1250) — compile-verified, matrix operand order checked by reading, not run on hardware
//
#include <hip/hip_runtime.h>
#include <hip/hip_bf16.h>

// ---------------- problem constants (from reference) ----------------
#define NUM_USERS 50000
#define NUM_ITEMS 50000
#define NN        (NUM_USERS + NUM_ITEMS)   // 100000
#define EMBED_K   64
#define FEAT      2048
#define N_EDGES   2000000
#define BATCH     8192
#define LEAKY     0.2f
#define N_LAYERS  3

// ---------------- WMMA types (CDNA5 / gfx1250, wave32) ----------------
typedef __attribute__((ext_vector_type(16))) __bf16 v16bf;
typedef __attribute__((ext_vector_type(8)))  float  v8f;

// 16 bf16 elements viewed as 8 packed dwords (element 2i -> low half of u[i]).
union BF16x16 {
    v16bf    v;
    unsigned u[8];
};

// Pack two f32 into one dword of two bf16 (lo -> [15:0], hi -> [31:16]).
// Round-half-up via +0x8000 bias, then one v_perm_b32 grabs both high halves.
__device__ __forceinline__ unsigned pk_bf16(float lo, float hi) {
#if __has_builtin(__builtin_amdgcn_cvt_pk_bf16_f32)
    typedef __attribute__((ext_vector_type(2))) __bf16 v2bf;
    v2bf p = __builtin_amdgcn_cvt_pk_bf16_f32(lo, hi);
    return __builtin_bit_cast(unsigned, p);
#else
    unsigned a = __builtin_bit_cast(unsigned, lo) + 0x8000u;
    unsigned b = __builtin_bit_cast(unsigned, hi) + 0x8000u;
    // {S0,S1}={b,a}: sel bytes 2,3 -> a[31:16]; sel bytes 6,7 -> b[31:16]
    return __builtin_amdgcn_perm(b, a, 0x07060302u);
#endif
}

// ---------------- kernel: pack f32 pairs -> bf16 dwords (weights, once) ------
__global__ void k_cvt_pk(const float* __restrict__ src, unsigned* __restrict__ dst,
                         int nPairs) {
    int i = blockIdx.x * blockDim.x + threadIdx.x;
    if (i < nPairs) {
        float2 f = *(const float2*)(src + 2 * i);
        dst[i] = pk_bf16(f.x, f.y);
    }
}

// ---------------- kernel: x = concat(Gu, Gi); acc = x ----------------
__global__ void k_init(const float* __restrict__ Gu, const float* __restrict__ Gi,
                       float* __restrict__ X, float* __restrict__ ACC,
                       int nuElems, int total) {
    int i = blockIdx.x * blockDim.x + threadIdx.x;
    if (i < total) {
        float v = (i < nuElems) ? Gu[i] : Gi[i - nuElems];
        X[i] = v;
        ACC[i] = v;
    }
}

__global__ void k_zero(float* __restrict__ p, int n) {
    int i = blockIdx.x * blockDim.x + threadIdx.x;
    if (i < n) p[i] = 0.0f;
}

// ---------------- kernel: edge scatter  agg[dst] += w * x[src] ----------------
// 16 threads per edge, 4 columns each -> 256B contiguous reads + atomics, L2-resident.
__global__ void k_scatter(const float* __restrict__ X,
                          const int* __restrict__ src, const int* __restrict__ dst,
                          const float* __restrict__ w, float* __restrict__ AGG, int E) {
    int gid = blockIdx.x * blockDim.x + threadIdx.x;
    int e = gid >> 4;
    if (e >= E) return;
    int c4 = (gid & 15) * 4;
    int s = src[e];
    int d = dst[e];
    float we = w[e];
    const float4 xv = *(const float4*)(X + (size_t)s * EMBED_K + c4);
    float* base = AGG + (size_t)d * EMBED_K + c4;
    atomicAdd(base + 0, xv.x * we);
    atomicAdd(base + 1, xv.y * we);
    atomicAdd(base + 2, xv.z * we);
    atomicAdd(base + 3, xv.w * we);
}

// B-tile fill from a pre-packed bf16 row (rowdw = dword pointer to row n).
// kbase in elements (even): elems 0..7 = K kbase.., elems 8..15 = K kbase+16..
__device__ __forceinline__ void load_btile(BF16x16& t, const unsigned* __restrict__ rowdw,
                                           int kbase) {
    uint4 q0 = *(const uint4*)(rowdw + (kbase >> 1));          // 16B: K kbase..+7
    uint4 q1 = *(const uint4*)(rowdw + ((kbase + 16) >> 1));   // 16B: K kbase+16..+23
    t.u[0] = q0.x; t.u[1] = q0.y; t.u[2] = q0.z; t.u[3] = q0.w;
    t.u[4] = q1.x; t.u[5] = q1.y; t.u[6] = q1.z; t.u[7] = q1.w;
}

// ---------------- kernel: per-layer dense GEMM via WMMA ----------------------
// One wave computes one 16-row tile of: H = (X+G)@W1^T + (X*G)@W2^T + b1 + b2.
// H aliases G (each wave reads/writes only its own 16 rows).
__global__ void k_layer(const float* __restrict__ X, const float* __restrict__ G,
                        float* __restrict__ H,
                        const unsigned* __restrict__ W1bf, const float* __restrict__ bias1,
                        const unsigned* __restrict__ W2bf, const float* __restrict__ bias2,
                        int numTiles) {
    int wave = (blockIdx.x * blockDim.x + threadIdx.x) >> 5;
    if (wave >= numTiles) return;                 // wave-uniform: EXEC stays all-ones
    int lane = threadIdx.x & 31;
    int row0 = wave * 16;

    int m  = lane & 15;
    int kb = (lane >> 4) ? 8 : 0;

    const float* xr = X + (size_t)(row0 + m) * EMBED_K;
    const float* gr = G + (size_t)(row0 + m) * EMBED_K;

    // A operands: (x+agg) and (x*agg), two K-halves each (K 0..31 / 32..63).
    BF16x16 As[2], Ap[2];
#pragma unroll
    for (int h = 0; h < 2; ++h) {
        int b0 = 32 * h + kb;
#pragma unroll
        for (int i = 0; i < 4; ++i) {
            float x0 = xr[b0 + 2 * i],      x1 = xr[b0 + 2 * i + 1];
            float g0 = gr[b0 + 2 * i],      g1 = gr[b0 + 2 * i + 1];
            float x2 = xr[b0 + 16 + 2 * i], x3 = xr[b0 + 16 + 2 * i + 1];
            float g2 = gr[b0 + 16 + 2 * i], g3 = gr[b0 + 16 + 2 * i + 1];
            As[h].u[i]     = pk_bf16(x0 + g0, x1 + g1);
            Ap[h].u[i]     = pk_bf16(x0 * g0, x1 * g1);
            As[h].u[4 + i] = pk_bf16(x2 + g2, x3 + g3);
            Ap[h].u[4 + i] = pk_bf16(x2 * g2, x3 * g3);
        }
    }

    int n = lane & 15;
    int moff = (lane >> 4) ? 8 : 0;

#pragma unroll
    for (int nt = 0; nt < 4; ++nt) {
        const unsigned* w1r = W1bf + (size_t)(nt * 16 + n) * (EMBED_K / 2);
        const unsigned* w2r = W2bf + (size_t)(nt * 16 + n) * (EMBED_K / 2);
        float bias = bias1[nt * 16 + n] + bias2[nt * 16 + n];
        v8f c;
#pragma unroll
        for (int r = 0; r < 8; ++r) c[r] = bias;

#pragma unroll
        for (int h = 0; h < 2; ++h) {
            BF16x16 B;
            load_btile(B, w1r, 32 * h + kb);
            c = __builtin_amdgcn_wmma_f32_16x16x32_bf16(false, As[h].v, false, B.v,
                                                        (short)0, c, false, false);
        }
#pragma unroll
        for (int h = 0; h < 2; ++h) {
            BF16x16 B;
            load_btile(B, w2r, 32 * h + kb);
            c = __builtin_amdgcn_wmma_f32_16x16x32_bf16(false, Ap[h].v, false, B.v,
                                                        (short)0, c, false, false);
        }

#pragma unroll
        for (int r = 0; r < 8; ++r)
            H[(size_t)(row0 + moff + r) * EMBED_K + nt * 16 + n] = c[r];
    }
}

// ---------------- kernel: leaky-relu + row L2-norm + layer-mean accumulate ----
// 64 threads per row, 4 rows per 256-thread block.
__global__ void k_norm_acc(float* __restrict__ H, float* __restrict__ ACC) {
    __shared__ float red[256];
    int t = threadIdx.x;
    int row = blockIdx.x * 4 + (t >> 6);
    int c = t & 63;
    float v = H[(size_t)row * EMBED_K + c];
    v = v > 0.0f ? v : LEAKY * v;
    red[t] = v * v;
    __syncthreads();
#pragma unroll
    for (int off = 32; off >= 1; off >>= 1) {
        if ((t & 63) < off) red[t] += red[t + off];
        __syncthreads();
    }
    float nrm = sqrtf(red[t & ~63]);
    float scale = 1.0f / fmaxf(nrm, 1e-12f);
    float o = v * scale;
    H[(size_t)row * EMBED_K + c] = o;
    ACC[(size_t)row * EMBED_K + c] += o;
}

// ---------------- kernel: projection GEMM  P = F[items] @ proj_w^T + b --------
// One wave per 16-batch-row tile; K=2048 in 32-wide bf16 WMMA steps.
__global__ void k_proj(const float* __restrict__ F, const unsigned* __restrict__ Wpbf,
                       const float* __restrict__ bp, const int* __restrict__ items,
                       float* __restrict__ P) {
    int wave = (blockIdx.x * blockDim.x + threadIdx.x) >> 5;
    if (wave >= BATCH / 16) return;               // wave-uniform
    int lane = threadIdx.x & 31;
    int row0 = wave * 16;

    int m  = lane & 15;
    int kb = (lane >> 4) ? 8 : 0;
    int item = items[row0 + m];
    const float* fr = F + (size_t)item * FEAT;
    int n = lane & 15;

    v8f acc[4];
#pragma unroll
    for (int nt = 0; nt < 4; ++nt)
#pragma unroll
        for (int r = 0; r < 8; ++r) acc[nt][r] = 0.0f;

    for (int k0 = 0; k0 < FEAT; k0 += 32) {
        BF16x16 a;
#pragma unroll
        for (int i = 0; i < 4; ++i) {
            int base = k0 + kb;
            a.u[i]     = pk_bf16(fr[base + 2 * i],      fr[base + 2 * i + 1]);
            a.u[4 + i] = pk_bf16(fr[base + 16 + 2 * i], fr[base + 16 + 2 * i + 1]);
        }
#pragma unroll
        for (int nt = 0; nt < 4; ++nt) {
            const unsigned* wr = Wpbf + (size_t)(nt * 16 + n) * (FEAT / 2);
            BF16x16 b;
            load_btile(b, wr, k0 + kb);
            acc[nt] = __builtin_amdgcn_wmma_f32_16x16x32_bf16(false, a.v, false, b.v,
                                                              (short)0, acc[nt], false, false);
        }
    }

    int moff = (lane >> 4) ? 8 : 0;
#pragma unroll
    for (int nt = 0; nt < 4; ++nt) {
        float bias = bp[nt * 16 + n];
#pragma unroll
        for (int r = 0; r < 8; ++r)
            P[(size_t)(row0 + moff + r) * EMBED_K + nt * 16 + n] = acc[nt][r] + bias;
    }
}

// ---------------- kernel: final dots:  xui = <gu,gi>/16 + <Tu, l2norm(P)> -----
__global__ void k_final(const float* __restrict__ ACC, const float* __restrict__ Tu,
                        const float* __restrict__ P, const int* __restrict__ users,
                        const int* __restrict__ items, float* __restrict__ out) {
    __shared__ float rg[256], rp[256], rt[256];
    int t = threadIdx.x;
    int b = blockIdx.x * 4 + (t >> 6);
    int c = t & 63;
    int u = users[b];
    int it = items[b];
    float gu = ACC[(size_t)u * EMBED_K + c];
    float gi = ACC[((size_t)NUM_USERS + it) * EMBED_K + c];
    float p  = P[(size_t)b * EMBED_K + c];
    float th = Tu[(size_t)u * EMBED_K + c];
    rg[t] = gu * gi;
    rp[t] = p * p;
    rt[t] = th * p;
    __syncthreads();
#pragma unroll
    for (int off = 32; off >= 1; off >>= 1) {
        if ((t & 63) < off) {
            rg[t] += rg[t + off];
            rp[t] += rp[t + off];
            rt[t] += rt[t + off];
        }
        __syncthreads();
    }
    if (c == 0) {
        float g   = rg[t];
        float nrm = fmaxf(sqrtf(rp[t]), 1e-12f);
        float td  = rt[t];
        out[b] = g * (1.0f / 16.0f) + td / nrm;  // /16 = mean-over-4 applied to both dots
    }
}

// ---------------- host launch ------------------------------------------------
extern "C" void kernel_launch(void* const* d_in, const int* in_sizes, int n_in,
                              void* d_out, int out_size, void* d_ws, size_t ws_size,
                              hipStream_t stream) {
    const float* Gu    = (const float*)d_in[0];
    const float* Gi    = (const float*)d_in[1];
    const float* Tu    = (const float*)d_in[2];
    const float* F     = (const float*)d_in[3];
    const float* projw = (const float*)d_in[4];
    const float* projb = (const float*)d_in[5];
    const float* W1    = (const float*)d_in[6];
    const float* b1    = (const float*)d_in[7];
    const float* W2    = (const float*)d_in[8];
    const float* b2    = (const float*)d_in[9];
    const float* ew    = (const float*)d_in[10];
    const int*   ei    = (const int*)d_in[11];
    const int*   users = (const int*)d_in[12];
    const int*   items = (const int*)d_in[13];
    float* out = (float*)d_out;

    const size_t NODE_ELEMS = (size_t)NN * EMBED_K;          // 6.4M floats
    float* X   = (float*)d_ws;                               // ping
    float* G   = X + NODE_ELEMS;                             // pong (agg / h)
    float* ACC = G + NODE_ELEMS;                             // sum of 4 embeddings
    float* P   = ACC + NODE_ELEMS;                           // BATCH x 64 projection
    unsigned* W1bf = (unsigned*)(P + (size_t)BATCH * EMBED_K);   // 3*64*32 dwords
    unsigned* W2bf = W1bf + N_LAYERS * EMBED_K * (EMBED_K / 2);
    unsigned* Wpbf = W2bf + N_LAYERS * EMBED_K * (EMBED_K / 2);  // 64*1024 dwords

    // ---- pre-pack all weights to bf16 (row-major, pairs along K) ----
    {
        int pw = N_LAYERS * EMBED_K * EMBED_K / 2;           // 6144 pairs each
        k_cvt_pk<<<(pw + 255) / 256, 256, 0, stream>>>(W1, W1bf, pw);
        k_cvt_pk<<<(pw + 255) / 256, 256, 0, stream>>>(W2, W2bf, pw);
        int pp = EMBED_K * FEAT / 2;                         // 65536 pairs
        k_cvt_pk<<<(pp + 255) / 256, 256, 0, stream>>>(projw, Wpbf, pp);
    }

    const int total = NN * EMBED_K;
    k_init<<<(total + 255) / 256, 256, 0, stream>>>(Gu, Gi, X, ACC, NUM_USERS * EMBED_K, total);

    const int tiles = NN / 16;                               // 6250
    for (int l = 0; l < 3; ++l) {
        k_zero<<<(total + 255) / 256, 256, 0, stream>>>(G, total);
        k_scatter<<<(N_EDGES * 16 + 255) / 256, 256, 0, stream>>>(
            X, ei, ei + N_EDGES, ew, G, N_EDGES);
        k_layer<<<(tiles + 3) / 4, 128, 0, stream>>>(
            X, G, G,
            W1bf + (size_t)l * EMBED_K * (EMBED_K / 2), b1 + (size_t)l * EMBED_K,
            W2bf + (size_t)l * EMBED_K * (EMBED_K / 2), b2 + (size_t)l * EMBED_K,
            tiles);
        k_norm_acc<<<NN / 4, 256, 0, stream>>>(G, ACC);
        float* tmp = X; X = G; G = tmp;                      // h becomes next layer's x
    }

    k_proj<<<((BATCH / 16) * 32 + 127) / 128, 128, 0, stream>>>(F, Wpbf, projb, items, P);
    k_final<<<BATCH / 4, 256, 0, stream>>>(ACC, Tu, P, users, items, out);
}
